// HybridModelImproved_14877766713462
// MI455X (gfx1250) — matrix operand order repaired and checked by reading, not compiled, not run on hardware
//
#include <hip/hip_runtime.h>

typedef __attribute__((ext_vector_type(16))) _Float16     v16h;
typedef __attribute__((ext_vector_type(8)))  float        v8f;
typedef __attribute__((ext_vector_type(8)))  unsigned int v8u;
typedef __attribute__((ext_vector_type(4)))  int          v4i;
typedef unsigned int u32;

// CDNA5 async global->LDS path (ASYNCcnt-tracked, bypasses VGPRs). Guarded so
// that toolchains without the builtins fall back to the plain copy loops.
// Probe-learned signature: (v4i AS1* gsrc, v4i AS3* ldst, imm offset, imm cpol)
#if defined(__AMDGCN__) &&                                                     \
    __has_builtin(__builtin_amdgcn_global_load_async_to_lds_b128) &&           \
    __has_builtin(__builtin_amdgcn_s_wait_asynccnt)
#define USE_ASYNC_LDS 1
typedef __attribute__((address_space(1))) v4i* gv4p;
typedef __attribute__((address_space(3))) v4i* lv4p;
__device__ __forceinline__ void async_b128(const void* g, void* l) {
  __builtin_amdgcn_global_load_async_to_lds_b128((gv4p)g, (lv4p)l, 0, 0);
}
#endif

// ---------------------------------------------------------------------------
// Kernel 0: weight repack (f32 -> f16, WMMA-B "k-pair packed" layouts)
//   conv2 im2col weights: k in [0,160) (pad from 150), n in [0,16)
//       w2p[ ((k>>1)*16 + n)*2 + (k&1) ]
//   fc1 transposed:       k in [0,256), n in [0,128) (pad from 120)
//       w1p[ ((k>>1)*128 + n)*2 + (k&1) ]
// ---------------------------------------------------------------------------
__global__ void prep_weights(const float* __restrict__ conv2_w,
                             const float* __restrict__ fc1_w,
                             _Float16* __restrict__ w2p,
                             _Float16* __restrict__ w1p) {
  int t = threadIdx.x;
  for (int i = t; i < 160 * 16; i += 256) {
    int k = i / 16, n = i % 16;
    float v = 0.f;
    if (k < 150) v = conv2_w[n * 150 + k];     // OIHW: o=n, (c,r,s)=k
    w2p[((k >> 1) * 16 + n) * 2 + (k & 1)] = (_Float16)v;
  }
  for (int i = t; i < 256 * 128; i += 256) {
    int k = i / 128, n = i % 128;
    float v = (n < 120) ? fc1_w[n * 256 + k] : 0.f;
    w1p[((k >> 1) * 128 + n) * 2 + (k & 1)] = (_Float16)v;
  }
}

// ---------------------------------------------------------------------------
// Kernel 1: conv1 (1->6, 5x5) + ReLU + 2x2 maxpool, fused. One thread per
// pooled output. relu(max(conv)+b) == max(relu(conv+b)) since bias is uniform
// over the pooling window and relu is monotone.
// act1 layout: [B][6][12][12] f16
// ---------------------------------------------------------------------------
__global__ void conv1_pool(const float* __restrict__ x,
                           const float* __restrict__ w,
                           const float* __restrict__ bias,
                           _Float16* __restrict__ act1, int B) {
  int idx = blockIdx.x * blockDim.x + threadIdx.x;
  if (idx >= B * 864) return;
  int px = idx % 12, py = (idx / 12) % 12, c = (idx / 144) % 6, b = idx / 864;
  const float* xb = x + (size_t)b * 784;
  const float* wc = w + c * 25;
  float wr[25];
#pragma unroll
  for (int i = 0; i < 25; ++i) wr[i] = wc[i];
  float mx = -1e30f;
#pragma unroll
  for (int dy = 0; dy < 2; ++dy)
#pragma unroll
    for (int dx = 0; dx < 2; ++dx) {
      int oy = 2 * py + dy, ox = 2 * px + dx;
      float acc = 0.f;
#pragma unroll
      for (int r = 0; r < 5; ++r)
#pragma unroll
        for (int s = 0; s < 5; ++s)
          acc = fmaf(xb[(oy + r) * 28 + ox + s], wr[r * 5 + s], acc);
      mx = fmaxf(mx, acc);
    }
  act1[idx] = (_Float16)fmaxf(0.f, mx + bias[c]);
}

// ---------------------------------------------------------------------------
// Kernel 2: conv2 as implicit GEMM on WMMA.
//   M = 64 output pixels (one image per block), N = 16 channels, K = 160
//   (padded from 6*5*5=150). 128 threads = 4 waves; wave w owns rows
//   16w..16w+15. 5 K-steps of v_wmma_f32_16x16x32_f16 per wave.
//   The whole 1728-B input image and the weight tile are staged to LDS via
//   the async global->LDS path; im2col then gathers purely from LDS (each
//   activation is reused ~12x by im2col). Epilogue: bias+ReLU+2x2 pool.
// act2 layout: [B][256] f16, flat index = ch*16 + py*4 + px (reference order)
// ---------------------------------------------------------------------------
#define C2K 160
__global__ void __launch_bounds__(128) conv2_wmma(
    const _Float16* __restrict__ act1, const _Float16* __restrict__ w2p,
    const float* __restrict__ conv2_b, _Float16* __restrict__ act2) {
  __shared__ __align__(16) _Float16 stage[864];     // one image [6][12][12]
  __shared__ __align__(16) _Float16 As[64 * C2K];   // im2col tile
  __shared__ __align__(16) _Float16 Bs[C2K * 16];   // k-pair packed weights
  __shared__ __align__(16) float    Cs[64 * 16];

  int b = blockIdx.x;
  int t = threadIdx.x;
  const _Float16* a1 = act1 + (size_t)b * 864;

#ifdef USE_ASYNC_LDS
  for (int i = t; i < 108; i += 128)                // 1728 B image -> LDS
    async_b128((const char*)a1 + i * 16, (char*)stage + i * 16);
  for (int i = t; i < 320; i += 128)                // 5120 B weights -> LDS
    async_b128((const char*)w2p + i * 16, (char*)Bs + i * 16);
  __builtin_amdgcn_s_wait_asynccnt(0);
#else
  for (int i = t; i < 864 / 2; i += 128)
    ((u32*)stage)[i] = ((const u32*)a1)[i];
  for (int i = t; i < C2K * 16 / 2; i += 128)
    ((u32*)Bs)[i] = ((const u32*)w2p)[i];
#endif
  __syncthreads();                                  // staging visible to WG

  for (int i = t; i < 64 * C2K; i += 128) {         // im2col from LDS stage
    int p = i / C2K, k = i % C2K;
    _Float16 v = (_Float16)0.f;
    if (k < 150) {
      int c = k / 25, rem = k % 25, r = rem / 5, s = rem % 5;
      int oy = p >> 3, ox = p & 7;
      v = stage[c * 144 + (oy + r) * 12 + (ox + s)];
    }
    As[i] = v;
  }
  __syncthreads();

  int wave = t >> 5, lane = t & 31;
  int hlf = lane >> 4, m = lane & 15;
  int row = wave * 16 + m;
  v8f acc = {};
#pragma unroll
  for (int kk = 0; kk < 5; ++kk) {
    int k0 = kk * 32;
    v8u au, bu;
#pragma unroll
    for (int q = 0; q < 8; ++q) {
      int ka = k0 + ((q < 4) ? (2 * q + 8 * hlf) : (16 + 2 * (q - 4) + 8 * hlf));
      au[q] = *(const u32*)&As[row * C2K + ka];
      int kb = k0 + ((q < 4) ? (2 * q + 8 * hlf) : (16 + 2 * (q - 4) + 8 * hlf));
      bu[q] = ((const u32*)Bs)[(kb >> 1) * 16 + m];
    }
    v16h a  = __builtin_bit_cast(v16h, au);
    v16h bb = __builtin_bit_cast(v16h, bu);
    acc = __builtin_amdgcn_wmma_f32_16x16x32_f16(false, a, false, bb,
                                                 (short)0, acc, false, false);
  }
#pragma unroll
  for (int j = 0; j < 8; ++j)                        // C: M=j+8*hlf, N=m
    Cs[(wave * 16 + j + 8 * hlf) * 16 + m] = acc[j];
  __syncthreads();

  for (int i = t; i < 256; i += 128) {               // bias+ReLU+maxpool
    int ch = i >> 4, py = (i >> 2) & 3, px = i & 3;
    float mx = -1e30f;
#pragma unroll
    for (int dy = 0; dy < 2; ++dy)
#pragma unroll
      for (int dx = 0; dx < 2; ++dx)
        mx = fmaxf(mx, Cs[((2 * py + dy) * 8 + (2 * px + dx)) * 16 + ch]);
    act2[(size_t)b * 256 + i] = (_Float16)fmaxf(0.f, mx + conv2_b[ch]);
  }
}

// ---------------------------------------------------------------------------
// Kernel 3: fc1 GEMM on WMMA. M=16 batch rows per block, N=128 (pad 120),
// K=256. 256 threads = 8 waves; wave w owns cols 16w..16w+15. 8 K-steps.
// A tile staged via async global->LDS; B fragments read straight from global
// (all 512 blocks share the same 64 KB -> L2/WGP$ hot).
// h1 layout: [B][128] f16
// ---------------------------------------------------------------------------
__global__ void __launch_bounds__(256) fc1_wmma(
    const _Float16* __restrict__ act2, const _Float16* __restrict__ w1p,
    const float* __restrict__ fc1_b, _Float16* __restrict__ h1) {
  __shared__ __align__(16) _Float16 As[16 * 256];
  int b0 = blockIdx.x * 16;
  int t = threadIdx.x;
#ifdef USE_ASYNC_LDS
  {
    const char* src = (const char*)(act2 + (size_t)b0 * 256);   // 8192 B
    for (int i = t; i < 512; i += 256)
      async_b128(src + i * 16, (char*)As + i * 16);
    __builtin_amdgcn_s_wait_asynccnt(0);
  }
#else
  for (int i = t; i < 16 * 256 / 2; i += 256)
    ((u32*)As)[i] = ((const u32*)(act2 + (size_t)b0 * 256))[i];
#endif
  __syncthreads();

  int wave = t >> 5, lane = t & 31;
  int hlf = lane >> 4, m = lane & 15;
  int ncol = wave * 16 + m;
  v8f acc = {};
#pragma unroll
  for (int kk = 0; kk < 8; ++kk) {
    int k0 = kk * 32;
    v8u au, bu;
#pragma unroll
    for (int q = 0; q < 8; ++q) {
      int ka = k0 + ((q < 4) ? (2 * q + 8 * hlf) : (16 + 2 * (q - 4) + 8 * hlf));
      au[q] = *(const u32*)&As[m * 256 + ka];
      int kb = k0 + ((q < 4) ? (2 * q + 8 * hlf) : (16 + 2 * (q - 4) + 8 * hlf));
      bu[q] = ((const u32*)w1p)[(kb >> 1) * 128 + ncol];
    }
    v16h a  = __builtin_bit_cast(v16h, au);
    v16h bb = __builtin_bit_cast(v16h, bu);
    acc = __builtin_amdgcn_wmma_f32_16x16x32_f16(false, a, false, bb,
                                                 (short)0, acc, false, false);
  }
  float bias = (ncol < 120) ? fc1_b[ncol] : 0.f;
#pragma unroll
  for (int j = 0; j < 8; ++j) {
    int brow = b0 + j + 8 * hlf;
    h1[(size_t)brow * 128 + ncol] = (_Float16)fmaxf(0.f, acc[j] + bias);
  }
}

// ---------------------------------------------------------------------------
// Kernel 4: fc2 + sigmoid + 4-qubit statevector sim + classifier.
// One thread per sample; 16 complex amplitudes held in registers, fully
// unrolled so all state indexing is constant-folded.
// ---------------------------------------------------------------------------
__device__ __forceinline__ float2 cmul(float2 a, float2 b) {
  return make_float2(a.x * b.x - a.y * b.y, a.x * b.y + a.y * b.x);
}
__device__ __forceinline__ float2 cadd(float2 a, float2 b) {
  return make_float2(a.x + b.x, a.y + b.y);
}
__device__ __forceinline__ void apply1q(float2* s, int wire, float2 g00,
                                        float2 g01, float2 g10, float2 g11) {
  int bit = 1 << (3 - wire);   // axis (wire+1) of [B,2,2,2,2] -> MSB-first bits
#pragma unroll
  for (int i = 0; i < 16; ++i) {
    if (!(i & bit)) {
      float2 a = s[i], b = s[i | bit];
      s[i]       = cadd(cmul(g00, a), cmul(g01, b));
      s[i | bit] = cadd(cmul(g10, a), cmul(g11, b));
    }
  }
}
__device__ __forceinline__ void cnot(float2* s, int ctrl, int tgt) {
  int bc = 1 << (3 - ctrl), bt = 1 << (3 - tgt);
#pragma unroll
  for (int i = 0; i < 16; ++i) {
    if ((i & bc) && !(i & bt)) {
      float2 tmp = s[i]; s[i] = s[i | bt]; s[i | bt] = tmp;
    }
  }
}

__global__ void tail_kernel(const _Float16* __restrict__ h1,
                            const float* __restrict__ fc2_w,
                            const float* __restrict__ fc2_b,
                            const float* __restrict__ qw,
                            const float* __restrict__ clf_w,
                            const float* __restrict__ clf_b,
                            float* __restrict__ out, int B) {
  int b = blockIdx.x * blockDim.x + threadIdx.x;
  if (b >= B) return;
  const _Float16* h = h1 + (size_t)b * 128;

  float ang[4];
#pragma unroll
  for (int j = 0; j < 4; ++j) {
    float acc = fc2_b[j];
    for (int k = 0; k < 120; ++k)
      acc = fmaf((float)h[k], fc2_w[j * 120 + k], acc);
    ang[j] = 3.14159265358979f / (1.f + __expf(-acc));   // pi * sigmoid
  }

  float2 s[16];
#pragma unroll
  for (int i = 0; i < 16; ++i) s[i] = make_float2(0.f, 0.f);
  s[0].x = 1.f;

#pragma unroll
  for (int w = 0; w < 4; ++w) {                          // RX embedding
    float c = __cosf(ang[w] * 0.5f), sn = __sinf(ang[w] * 0.5f);
    apply1q(s, w, make_float2(c, 0), make_float2(0, -sn),
                  make_float2(0, -sn), make_float2(c, 0));
  }
#pragma unroll
  for (int l = 0; l < 2; ++l) {                          // entangling layers
#pragma unroll
    for (int w = 0; w < 4; ++w) {
      float phi = qw[(l * 4 + w) * 3 + 0], th = qw[(l * 4 + w) * 3 + 1],
            om  = qw[(l * 4 + w) * 3 + 2];
      float ct = __cosf(th * 0.5f), st = __sinf(th * 0.5f);
      float pa = 0.5f * (phi + om), pb = 0.5f * (phi - om);
      apply1q(s, w,
              make_float2( __cosf(pa) * ct, -__sinf(pa) * ct),
              make_float2(-__cosf(pb) * st, -__sinf(pb) * st),
              make_float2( __cosf(pb) * st, -__sinf(pb) * st),
              make_float2( __cosf(pa) * ct,  __sinf(pa) * ct));
    }
    int r = (l % 3) + 1;
#pragma unroll
    for (int w = 0; w < 4; ++w) cnot(s, w, (w + r) & 3);
  }

  float z[4] = {0.f, 0.f, 0.f, 0.f};
#pragma unroll
  for (int i = 0; i < 16; ++i) {
    float p = s[i].x * s[i].x + s[i].y * s[i].y;
#pragma unroll
    for (int w = 0; w < 4; ++w) z[w] += ((i >> (3 - w)) & 1) ? -p : p;
  }
#pragma unroll
  for (int o = 0; o < 10; ++o) {
    float acc = clf_b[o];
#pragma unroll
    for (int j = 0; j < 4; ++j) acc = fmaf(z[j], clf_w[o * 4 + j], acc);
    out[(size_t)b * 10 + o] = acc;
  }
}

// ---------------------------------------------------------------------------
extern "C" void kernel_launch(void* const* d_in, const int* in_sizes, int n_in,
                              void* d_out, int out_size, void* d_ws,
                              size_t ws_size, hipStream_t stream) {
  const float* x       = (const float*)d_in[0];
  const float* conv1_w = (const float*)d_in[1];
  const float* conv1_b = (const float*)d_in[2];
  const float* conv2_w = (const float*)d_in[3];
  const float* conv2_b = (const float*)d_in[4];
  const float* fc1_w   = (const float*)d_in[5];
  const float* fc1_b   = (const float*)d_in[6];
  const float* fc2_w   = (const float*)d_in[7];
  const float* fc2_b   = (const float*)d_in[8];
  const float* qw      = (const float*)d_in[9];
  const float* clf_w   = (const float*)d_in[10];
  const float* clf_b   = (const float*)d_in[11];
  float* out = (float*)d_out;

  int B = in_sizes[0] / 784;

  char* ws = (char*)d_ws;
  size_t W2OFF = 0;                                   // 160*16 f16   = 5120 B
  size_t W1OFF = 5120;                                // 256*128 f16  = 65536 B
  size_t A1OFF = 70656;                               // B*864 f16
  size_t A2OFF = A1OFF + (size_t)B * 864 * 2;         // B*256 f16
  size_t H1OFF = A2OFF + (size_t)B * 256 * 2;         // B*128 f16
  _Float16* w2p  = (_Float16*)(ws + W2OFF);
  _Float16* w1p  = (_Float16*)(ws + W1OFF);
  _Float16* act1 = (_Float16*)(ws + A1OFF);
  _Float16* act2 = (_Float16*)(ws + A2OFF);
  _Float16* h1   = (_Float16*)(ws + H1OFF);

  prep_weights<<<1, 256, 0, stream>>>(conv2_w, fc1_w, w2p, w1p);
  conv1_pool<<<(B * 864 + 255) / 256, 256, 0, stream>>>(x, conv1_w, conv1_b,
                                                        act1, B);
  conv2_wmma<<<B, 128, 0, stream>>>(act1, w2p, conv2_b, act2);
  fc1_wmma<<<B / 16, 256, 0, stream>>>(act2, w1p, fc1_b, h1);
  tail_kernel<<<(B + 255) / 256, 256, 0, stream>>>(h1, fc2_w, fc2_b, qw,
                                                   clf_w, clf_b, out, B);
}
